// MogLSTM_7464653160804
// MI455X (gfx1250) — matrix operand order repaired and checked by reading, not compiled, not run on hardware
//
#include <hip/hip_runtime.h>
#include <hip/hip_bf16.h>

// ---------------------------------------------------------------------------
// MogLSTM persistent kernel for gfx1250 (MI455X).
// bf16 WMMA (v_wmma_f32_16x16x32_bf16) GEMMs, f32 accumulate.
// - weights converted once to bf16, L2-resident (13.6 MB << 192 MB L2),
//   loaded non-temporally (no WGP$ pollution, zero intra-WGP reuse)
// - per-block A strips staged in LDS (8 waves share one 16-row strip),
//   padded stride for conflict-minimal ds_load_b128 fragment reads
// - single persistent grid, L2 atomic-counter barrier for the recurrence
// ---------------------------------------------------------------------------

#define T_SEQ    1024
#define BATCH    64
#define HID      512
#define GATESZ   (4 * HID)
#define BH       (BATCH * HID)
#define NWG      64
#define TPB      256
#define NTHREADS (NWG * TPB)
#define LDSTRIDE 520   // 512 + 8 shorts: 1040B rows -> lanes spread 4 banks apart

typedef __attribute__((ext_vector_type(16))) __bf16       v16bf;
typedef __attribute__((ext_vector_type(8)))  float        v8f;
typedef __attribute__((ext_vector_type(4)))  unsigned int v4u;

struct Frag32B { v4u lo, hi; };

__device__ __forceinline__ v16bf frag_cast(v4u lo, v4u hi) {
    Frag32B f; f.lo = lo; f.hi = hi;
    return __builtin_bit_cast(v16bf, f);
}

// f32 -> bf16 round-to-nearest-even
__device__ __forceinline__ unsigned short f2bf(float f) {
    unsigned u = __float_as_uint(f);
    u += 0x7FFFu + ((u >> 16) & 1u);
    return (unsigned short)(u >> 16);
}

__device__ __forceinline__ float sigmoidf_(float x) {
    return 1.0f / (1.0f + __expf(-x));
}
__device__ __forceinline__ float tanhf_(float x) {
    float e = __expf(2.0f * x);
    return 1.0f - 2.0f / (e + 1.0f);
}

// Stage one 16-row x 512-col bf16 strip into LDS (padded stride).
// 1024 x 16B chunks over 256 threads; both sides 16B aligned.
__device__ __forceinline__ void stage_a(const unsigned short* __restrict__ src,
                                        int row0, unsigned short* lds) {
    for (int j = threadIdx.x; j < 16 * 64; j += TPB) {
        const int row = j >> 6;
        const int c16 = j & 63;
        *(v4u*)(lds + row * LDSTRIDE + c16 * 8) =
            *(const v4u*)(src + (size_t)(row0 + row) * HID + c16 * 8);
    }
}

// A fragment (16x32 bf16) from the LDS strip.
// Lane L: m = L&15, hg = L>>4; elems 0..15 -> K = k0+hg*16+e (contiguous).
__device__ __forceinline__ v16bf load_a_lds(const unsigned short* lds,
                                            int k0, int lane) {
    const int m  = lane & 15;
    const int hg = lane >> 4;
    const unsigned short* p = lds + m * LDSTRIDE + k0 + hg * 16;
    v4u lo = *(const v4u*)(p);
    v4u hi = *(const v4u*)(p + 8);
    return frag_cast(lo, hi);
}

// B fragment: 32x16 bf16 of W^T where W is [N, K] row-major (torch layout).
// B[k][n] = W[n][k]. Lane L: n = L&15, K = (L>>4)*16 + e -> contiguous 32B.
// Non-temporal: weights stream with no near-cache reuse (L2-resident only).
__device__ __forceinline__ v16bf load_b_nt(const unsigned short* __restrict__ w,
                                           int col0, int k0, int lane) {
    const int n  = lane & 15;
    const int hg = lane >> 4;
    const unsigned short* p = w + (size_t)(col0 + n) * HID + k0 + hg * 16;
    v4u lo = __builtin_nontemporal_load((const v4u*)(p));
    v4u hi = __builtin_nontemporal_load((const v4u*)(p + 8));
    return frag_cast(lo, hi);
}

__device__ __forceinline__ v8f wmma_bf16(v16bf a, v16bf b, v8f c) {
    return __builtin_amdgcn_wmma_f32_16x16x32_bf16(
        /*neg_a=*/false, a, /*neg_b=*/false, b,
        /*c_mod=*/(short)0, c, /*reuse_a=*/false, /*reuse_b=*/false);
}

// Device-wide barrier: monotonic L2 atomic counter (reset per launch).
__device__ __forceinline__ void grid_sync(unsigned* bar) {
    __syncthreads();
    if (threadIdx.x == 0) {
        __threadfence();
        unsigned ticket = atomicAdd(bar, 1u);
        unsigned target = (ticket / NWG + 1u) * NWG;
        while (atomicAdd(bar, 0u) < target) {
            __builtin_amdgcn_s_sleep(1);
        }
        __threadfence();
    }
    __syncthreads();
}

// ---- phase: one mogrifier step, fused GEMM + 2*sigmoid gating -------------
// tmp = src_bf @ W^T + bias ; tgt_out = 2*sigmoid(tmp) * tgt_in
// Active blocks: 16 (128 tiles, mt uniform per block).
__device__ void mog_phase(const unsigned short* __restrict__ src_bf,
                          const unsigned short* __restrict__ W,
                          const float* __restrict__ bias,
                          const float* __restrict__ tgt_in,
                          float* __restrict__ tgt_out_f,
                          unsigned short* __restrict__ tgt_out_bf,
                          int lane, unsigned short* lds) {
    const int wg = blockIdx.x;
    if (wg >= (BATCH / 16) * 4) return;                 // blocks 0..15
    const int mt = wg >> 2;                             // uniform per block
    const int nt = ((wg & 3) << 3) + (threadIdx.x >> 5);
    const int m0 = mt * 16, n0 = nt * 16;

    stage_a(src_bf, m0, lds);
    __syncthreads();

    v8f acc = {};
    for (int k = 0; k < HID; k += 32) {
        v16bf a = load_a_lds(lds, k, lane);
        v16bf b = load_b_nt(W, n0, k, lane);
        acc = wmma_bf16(a, b, acc);
    }
    const int n  = n0 + (lane & 15);
    const int mb = m0 + 8 * (lane >> 4);
    const float bn = bias[n];
#pragma unroll
    for (int r = 0; r < 8; ++r) {
        const int idx = (mb + r) * HID + n;
        float g  = 2.0f * sigmoidf_(acc[r] + bn);
        float nv = g * tgt_in[idx];
        tgt_out_f[idx]  = nv;
        tgt_out_bf[idx] = f2bf(nv);
    }
    // LDS safe to reuse after the grid_sync that follows (it begins with
    // __syncthreads, so all waves finish reading before re-staging).
}

// ---- phase: LSTM gate GEMMs, g = x@Wih^T + h@Whh^T + bih + bhh ------------
// 512 tiles over 512 waves; mt uniform per block -> stage x and h strips.
__device__ void gates_phase(const unsigned short* __restrict__ x_bf,
                            const unsigned short* __restrict__ h_bf,
                            const unsigned short* __restrict__ Wih,
                            const unsigned short* __restrict__ Whh,
                            const float* __restrict__ bih,
                            const float* __restrict__ bhh,
                            float* __restrict__ gates,
                            int lane, unsigned short* lds_x,
                            unsigned short* lds_h) {
    const int wg = blockIdx.x;
    const int mt = wg >> 4;                              // uniform per block
    const int nt = ((wg & 15) << 3) + (threadIdx.x >> 5);
    const int m0 = mt * 16, n0 = nt * 16;

    stage_a(x_bf, m0, lds_x);
    stage_a(h_bf, m0, lds_h);
    __syncthreads();

    v8f acc = {};
    for (int k = 0; k < HID; k += 32) {
        v16bf a = load_a_lds(lds_x, k, lane);
        v16bf b = load_b_nt(Wih, n0, k, lane);
        acc = wmma_bf16(a, b, acc);
    }
    for (int k = 0; k < HID; k += 32) {
        v16bf a = load_a_lds(lds_h, k, lane);
        v16bf b = load_b_nt(Whh, n0, k, lane);
        acc = wmma_bf16(a, b, acc);
    }
    const int n  = n0 + (lane & 15);
    const int mb = m0 + 8 * (lane >> 4);
    const float bn = bih[n] + bhh[n];
#pragma unroll
    for (int r = 0; r < 8; ++r)
        gates[(size_t)(mb + r) * GATESZ + n] = acc[r] + bn;
}

// ---- phase: elementwise LSTM cell update ----------------------------------
__device__ void lstm_elem(const float* __restrict__ gates,
                          float* __restrict__ hf, float* __restrict__ cf,
                          unsigned short* __restrict__ hbf,
                          int layer, int t, float* __restrict__ out, int tid) {
    for (int idx = tid; idx < BH; idx += NTHREADS) {
        const int m = idx >> 9;           // / HID
        const int n = idx & (HID - 1);
        const float* g = gates + (size_t)m * GATESZ;
        float ig = sigmoidf_(g[n]);
        float fg = sigmoidf_(g[HID + n]);
        float gg = tanhf_(g[2 * HID + n]);
        float og = sigmoidf_(g[3 * HID + n]);
        float c  = fg * cf[idx] + ig * gg;
        float h  = og * tanhf_(c);
        cf[idx]  = c;
        hf[idx]  = h;
        hbf[idx] = f2bf(h);
        if (layer == 1) out[(size_t)t * BH + idx] = h;
        if (t == T_SEQ - 1) {
            out[(size_t)T_SEQ * BH + (size_t)layer * BH + idx] = h;                  // last_hidden
            out[(size_t)T_SEQ * BH + 2 * (size_t)BH + (size_t)layer * BH + idx] = c; // last_cell
        }
    }
}

// ---- setup kernels --------------------------------------------------------
__global__ void cvt_bf16_kernel(const float* __restrict__ src,
                                unsigned short* __restrict__ dst, int n) {
    int i = blockIdx.x * blockDim.x + threadIdx.x;
    const int stride = gridDim.x * blockDim.x;
    for (; i < n; i += stride) dst[i] = f2bf(src[i]);
}

__global__ void init_bar_kernel(unsigned* bar) {
    if (threadIdx.x < 64) bar[threadIdx.x] = 0u;
}

// ---- persistent recurrence kernel -----------------------------------------
__global__ void __launch_bounds__(TPB)
moglstm_kernel(const float* __restrict__ input_seq,
               const float* __restrict__ mogb,
               const float* __restrict__ b_ih,
               const float* __restrict__ b_hh,
               const unsigned short* __restrict__ mogW_bf,
               const unsigned short* __restrict__ Wih_bf,
               const unsigned short* __restrict__ Whh_bf,
               float* __restrict__ xl_f, float* __restrict__ hl_f,
               unsigned short* __restrict__ xl_bf, unsigned short* __restrict__ hl_bf,
               float* __restrict__ h_f, float* __restrict__ c_f,
               unsigned short* __restrict__ h_bf,
               float* __restrict__ gates_f,
               unsigned* __restrict__ bar,
               float* __restrict__ out) {
    __shared__ unsigned short lds_x[16 * LDSTRIDE];   // 16.6 KB
    __shared__ unsigned short lds_h[16 * LDSTRIDE];   // 16.6 KB (of 320 KB WGP LDS)

    const int tid  = blockIdx.x * TPB + threadIdx.x;
    const int lane = threadIdx.x & 31;

    // zero carry state (h1,c1,h2,c2 and bf16 mirrors)
    for (int i = tid; i < 2 * BH; i += NTHREADS) {
        h_f[i] = 0.0f; c_f[i] = 0.0f; h_bf[i] = 0;
    }
    grid_sync(bar);

    for (int t = 0; t < T_SEQ; ++t) {
        for (int layer = 0; layer < 2; ++layer) {
            // cell input x: layer 0 -> input_seq[t]; layer 1 -> fresh h1 (h_f[0])
            const float* x_src = layer ? h_f : (input_seq + (size_t)t * BH);
            const unsigned short* hcarry_bf = h_bf + (size_t)layer * BH;
            const float*          hcarry_f  = h_f + (size_t)layer * BH;

            // 5 mogrifier steps (i=0,2,4 update x; i=1,3 update h)
            for (int i = 0; i < 5; ++i) {
                const unsigned short* W = mogW_bf + (size_t)(layer * 5 + i) * HID * HID;
                const float* bias = mogb + (size_t)(layer * 5 + i) * HID;
                if ((i & 1) == 0) {
                    mog_phase(i == 0 ? hcarry_bf : hl_bf, W, bias,
                              i == 0 ? x_src : xl_f,
                              xl_f, xl_bf, lane, lds_x);
                } else {
                    mog_phase(xl_bf, W, bias,
                              i == 1 ? hcarry_f : hl_f,
                              hl_f, hl_bf, lane, lds_x);
                }
                grid_sync(bar);
            }

            gates_phase(xl_bf, hl_bf,
                        Wih_bf + (size_t)layer * GATESZ * HID,
                        Whh_bf + (size_t)layer * GATESZ * HID,
                        b_ih + (size_t)layer * GATESZ,
                        b_hh + (size_t)layer * GATESZ,
                        gates_f, lane, lds_x, lds_h);
            grid_sync(bar);

            lstm_elem(gates_f,
                      h_f + (size_t)layer * BH,
                      c_f + (size_t)layer * BH,
                      h_bf + (size_t)layer * BH,
                      layer, t, out, tid);
            grid_sync(bar);
        }
    }
}

// ---------------------------------------------------------------------------
extern "C" void kernel_launch(void* const* d_in, const int* in_sizes, int n_in,
                              void* d_out, int out_size, void* d_ws, size_t ws_size,
                              hipStream_t stream) {
    (void)in_sizes; (void)n_in; (void)out_size; (void)ws_size;
    const float* input_seq = (const float*)d_in[0];   // [T,B,H]
    const float* mogW      = (const float*)d_in[1];   // [2,5,H,H]
    const float* mogb      = (const float*)d_in[2];   // [2,5,H]
    const float* W_ih      = (const float*)d_in[3];   // [2,4H,H]
    const float* W_hh      = (const float*)d_in[4];   // [2,4H,H]
    const float* b_ih      = (const float*)d_in[5];   // [2,4H]
    const float* b_hh      = (const float*)d_in[6];   // [2,4H]

    char* ws = (char*)d_ws;
    // workspace layout (all offsets 1KB-aligned)
    unsigned*       bar      = (unsigned*)(ws + 0);                    // 256 B
    unsigned short* mogW_bf  = (unsigned short*)(ws + 1024);           // 5,242,880 B
    unsigned short* Wih_bf   = (unsigned short*)(ws + 5243904);        // 4,194,304 B
    unsigned short* Whh_bf   = (unsigned short*)(ws + 9438208);        // 4,194,304 B
    float*          xl_f     = (float*)(ws + 13632512);                // 131,072 B
    float*          hl_f     = (float*)(ws + 13763584);                // 131,072 B
    unsigned short* xl_bf    = (unsigned short*)(ws + 13894656);       //  65,536 B
    unsigned short* hl_bf    = (unsigned short*)(ws + 13960192);       //  65,536 B
    float*          h_f      = (float*)(ws + 14025728);                // 262,144 B
    float*          c_f      = (float*)(ws + 14287872);                // 262,144 B
    unsigned short* h_bf     = (unsigned short*)(ws + 14550016);       // 131,072 B
    float*          gates_f  = (float*)(ws + 14681088);                // 524,288 B
    // total: ~14.5 MB

    init_bar_kernel<<<1, 64, 0, stream>>>(bar);

    const int nMog = 2 * 5 * HID * HID;    // 2,621,440
    const int nIH  = 2 * GATESZ * HID;     // 2,097,152
    cvt_bf16_kernel<<<1024, 256, 0, stream>>>(mogW, mogW_bf, nMog);
    cvt_bf16_kernel<<<1024, 256, 0, stream>>>(W_ih, Wih_bf, nIH);
    cvt_bf16_kernel<<<1024, 256, 0, stream>>>(W_hh, Whh_bf, nIH);

    moglstm_kernel<<<NWG, TPB, 0, stream>>>(
        input_seq, mogb, b_ih, b_hh,
        mogW_bf, Wih_bf, Whh_bf,
        xl_f, hl_f, xl_bf, hl_bf,
        h_f, c_f, h_bf, gates_f,
        bar, (float*)d_out);
}
